// BitLinear_74552042324361
// MI455X (gfx1250) — matrix operand order repaired and checked by reading, not compiled, not run on hardware
//
#include <hip/hip_runtime.h>

typedef _Float16 v16h __attribute__((ext_vector_type(16)));
typedef _Float16 v8h  __attribute__((ext_vector_type(8)));
typedef _Float16 v4h  __attribute__((ext_vector_type(4)));
typedef float    v8f  __attribute__((ext_vector_type(8)));

#define NBITS 8

// ---------------------------------------------------------------------------
// CDNA5 async global->LDS copy (16 B per lane), tracked by ASYNCcnt.
// ---------------------------------------------------------------------------
typedef __attribute__((address_space(3))) char lds_char_t;

__device__ __forceinline__ void async_copy_b128(void* lds_dst, const void* gsrc) {
    lds_char_t* lp = (lds_char_t*)(unsigned long long)lds_dst;  // low 32 bits = LDS offset
    asm volatile("global_load_async_to_lds_b128 %0, %1, off"
                 :
                 : "v"(lp), "v"(gsrc)
                 : "memory");
}

#define WAIT_ASYNCCNT(n) asm volatile("s_wait_asynccnt %0" :: "n"(n) : "memory")

// ---------------------------------------------------------------------------
// Phase 1a: decode bitplanes -> f16 weight  W[o][i] = code/255 * sign * scale
// ---------------------------------------------------------------------------
__global__ void decode_weight_kernel(const float* __restrict__ bw,
                                     const float* __restrict__ wsign,
                                     const float* __restrict__ scale,
                                     _Float16* __restrict__ Wh,
                                     int total) {
    int idx = blockIdx.x * blockDim.x + threadIdx.x;
    if (idx >= total) return;
    const float4* p = (const float4*)(bw + (size_t)idx * NBITS);
    float4 b0 = p[0];
    float4 b1 = p[1];
    float code = b0.x * 128.f + b0.y * 64.f + b0.z * 32.f + b0.w * 16.f
               + b1.x * 8.f   + b1.y * 4.f  + b1.z * 2.f  + b1.w;
    float s   = wsign[idx];
    float sgn = (s > 0.f) ? 1.f : (s < 0.f ? -1.f : 0.f);
    Wh[idx] = (_Float16)(code * (1.f / 255.f) * sgn * scale[0]);
}

// ---------------------------------------------------------------------------
// Phase 1b: x f32 -> f16 (vectorized by 4)
// ---------------------------------------------------------------------------
__global__ void convert_x_kernel(const float* __restrict__ x,
                                 _Float16* __restrict__ xh,
                                 int total4) {
    int idx = blockIdx.x * blockDim.x + threadIdx.x;
    if (idx >= total4) return;
    float4 v = ((const float4*)x)[idx];
    v4h h = { (_Float16)v.x, (_Float16)v.y, (_Float16)v.z, (_Float16)v.w };
    ((v4h*)xh)[idx] = h;
}

// ---------------------------------------------------------------------------
// Phase 1c: decode bias -> f32
// ---------------------------------------------------------------------------
__global__ void decode_bias_kernel(const float* __restrict__ bb,
                                   const float* __restrict__ bs,
                                   const float* __restrict__ bscale,
                                   float* __restrict__ bias,
                                   int OUT) {
    int o = blockIdx.x * blockDim.x + threadIdx.x;
    if (o >= OUT) return;
    const float4* p = (const float4*)(bb + (size_t)o * NBITS);
    float4 b0 = p[0];
    float4 b1 = p[1];
    float code = b0.x * 128.f + b0.y * 64.f + b0.z * 32.f + b0.w * 16.f
               + b1.x * 8.f   + b1.y * 4.f  + b1.z * 2.f  + b1.w;
    float s   = bs[o];
    float sgn = (s > 0.f) ? 1.f : (s < 0.f ? -1.f : 0.f);
    bias[o] = code * (1.f / 255.f) * sgn * bscale[0];
}

// ---------------------------------------------------------------------------
// Phase 2: y = xh @ Wh^T + bias.
// Block = 256 threads = 8 waves (2M x 4N), wave tile 64x64 (4x4 WMMA accs).
// Block tile 128(M) x 256(N), K-step 32.
//
// Pipeline per k-step:
//   - all waves issue 6 async b128 global->LDS copies for step i+1 (ASYNCcnt)
//   - s_wait_asynccnt 6  (step i's copies done), barrier
//   - 16 ds_load_b128 fragment reads + 16 v_wmma_f32_16x16x32_f16
//   - barrier (buffer consumed before step i+2 overwrites it)
// LDS: double-buffered A(128x32 f16) + B(256x32 f16), rows padded to 80 B
// for conflict-minimal b128 reads. 2 x 30720 B = 60 KB.
//
// Fragment layouts (wave32, ISA 7.12.2), row = tile row, bytes within row:
//   A: VGPRs 0-3 = bytes [half*16, +16), VGPRs 4-7 = bytes [32+half*16, +16)
//   B: VGPRs 0-3 = bytes [half*16, +16), VGPRs 4-7 = bytes [32+half*16, +16) (Wh is N-major)
//   C/D: VGPR r -> row half*8 + r, lane l16 -> col
// ---------------------------------------------------------------------------
#define LDA_ROW 80
#define LDB_ROW 80
#define ABUF_BYTES (128 * LDA_ROW)           // 10240
#define BBUF_BYTES (256 * LDB_ROW)           // 20480
#define BUF_BYTES  (ABUF_BYTES + BBUF_BYTES) // 30720

__global__ __launch_bounds__(256)
void bitlinear_wmma_kernel(const _Float16* __restrict__ A,  // xh [M][K]
                           const _Float16* __restrict__ B,  // Wh [N][K]
                           const float* __restrict__ bias,  // [N]
                           float* __restrict__ out,         // [M][N]
                           int M, int N, int K) {
    __shared__ __align__(16) char smem[2 * BUF_BYTES];

    const int t    = threadIdx.x;
    const int lane = t & 31;
    const int wave = t >> 5;             // 0..7
    const int l16  = lane & 15;
    const int half = (lane >> 4) & 1;
    const int wm   = wave >> 2;          // 0..1
    const int wn   = wave & 3;           // 0..3

    const int mBlock = blockIdx.y * 128;
    const int nBlock = blockIdx.x * 256;

    // Issue one k-step's tile copies: A = 512 chunks of 16 B, B = 1024 chunks.
    // Thread t owns chunks {t, t+256} of A and {t, t+256, t+512, t+768} of B.
    auto stage = [&](int buf, int k0) {
        char* base = smem + buf * BUF_BYTES;
#pragma unroll
        for (int j = 0; j < 2; ++j) {
            int c = t + j * 256;
            int row = c >> 2, seg = c & 3;
            const char* g = (const char*)(A + (size_t)(mBlock + row) * K + k0) + seg * 16;
            async_copy_b128(base + row * LDA_ROW + seg * 16, g);
        }
#pragma unroll
        for (int j = 0; j < 4; ++j) {
            int c = t + j * 256;
            int row = c >> 2, seg = c & 3;
            const char* g = (const char*)(B + (size_t)(nBlock + row) * K + k0) + seg * 16;
            async_copy_b128(base + ABUF_BYTES + row * LDB_ROW + seg * 16, g);
        }
    };

    v8f acc[4][4];
#pragma unroll
    for (int mi = 0; mi < 4; ++mi)
#pragma unroll
        for (int ni = 0; ni < 4; ++ni)
            acc[mi][ni] = (v8f){0.f, 0.f, 0.f, 0.f, 0.f, 0.f, 0.f, 0.f};

    const int steps = K / 32;
    stage(0, 0);

    for (int i = 0; i < steps; ++i) {
        const int cur = i & 1;
        if (i + 1 < steps) {
            stage(cur ^ 1, (i + 1) * 32);
            WAIT_ASYNCCNT(6);   // the 6 just-issued remain; step i's copies done
        } else {
            WAIT_ASYNCCNT(0);
        }
        __syncthreads();        // all waves' copies for buf[cur] visible

        const char* aBase = smem + cur * BUF_BYTES;
        const char* bBase = aBase + ABUF_BYTES;

        v16h af[4];
        v16h bf[4];
#pragma unroll
        for (int mi = 0; mi < 4; ++mi) {
            const char* r = aBase + (wm * 64 + mi * 16 + l16) * LDA_ROW;
            v8h lo = *(const v8h*)(r + half * 16);
            v8h hi = *(const v8h*)(r + 32 + half * 16);
#pragma unroll
            for (int q = 0; q < 8; ++q) { af[mi][q] = lo[q]; af[mi][q + 8] = hi[q]; }
        }
#pragma unroll
        for (int ni = 0; ni < 4; ++ni) {
            const char* r = bBase + (wn * 64 + ni * 16 + l16) * LDB_ROW;
            v8h lo = *(const v8h*)(r + half * 16);
            v8h hi = *(const v8h*)(r + 32 + half * 16);
#pragma unroll
            for (int q = 0; q < 8; ++q) { bf[ni][q] = lo[q]; bf[ni][q + 8] = hi[q]; }
        }

#pragma unroll
        for (int mi = 0; mi < 4; ++mi)
#pragma unroll
            for (int ni = 0; ni < 4; ++ni)
                acc[mi][ni] = __builtin_amdgcn_wmma_f32_16x16x32_f16(
                    false, af[mi], false, bf[ni], (short)0, acc[mi][ni],
                    false, false);

        __syncthreads();        // buf[cur] fully consumed before being restaged
    }

#pragma unroll
    for (int mi = 0; mi < 4; ++mi) {
#pragma unroll
        for (int ni = 0; ni < 4; ++ni) {
            const int col = nBlock + wn * 64 + ni * 16 + l16;
            const float bv = bias[col];
#pragma unroll
            for (int r = 0; r < 8; ++r) {
                const int row = mBlock + wm * 64 + mi * 16 + half * 8 + r;
                out[(size_t)row * N + col] = acc[mi][ni][r] + bv;
            }
        }
    }
}

// ---------------------------------------------------------------------------
// Host launcher
// ---------------------------------------------------------------------------
extern "C" void kernel_launch(void* const* d_in, const int* in_sizes, int n_in,
                              void* d_out, int out_size, void* d_ws, size_t ws_size,
                              hipStream_t stream) {
    (void)n_in; (void)out_size; (void)ws_size;

    const float* x         = (const float*)d_in[0];  // [B,S,IN]
    const float* bweight   = (const float*)d_in[1];  // [OUT,IN,8]
    const float* wsign     = (const float*)d_in[2];  // [OUT,IN]
    const float* scale     = (const float*)d_in[3];  // [1]
    const float* bbias     = (const float*)d_in[4];  // [OUT,8]
    const float* bsign     = (const float*)d_in[5];  // [OUT]
    const float* biasscale = (const float*)d_in[6];  // [1]
    float* out = (float*)d_out;

    const int OUT = in_sizes[5];
    const int IN  = in_sizes[2] / OUT;
    const int M   = in_sizes[0] / IN;  // B*S
    const int N   = OUT;
    const int K   = IN;

    char* ws = (char*)d_ws;
    _Float16* xh   = (_Float16*)ws;                                   // M*K f16
    _Float16* Wh   = (_Float16*)(ws + (size_t)M * K * 2);             // N*K f16
    float*    bias = (float*)(ws + (size_t)M * K * 2 + (size_t)N * K * 2);

    {   // decode weights
        int total = N * K;
        decode_weight_kernel<<<(total + 255) / 256, 256, 0, stream>>>(
            bweight, wsign, scale, Wh, total);
    }
    {   // convert activations to f16
        int total4 = (M * K) / 4;
        convert_x_kernel<<<(total4 + 255) / 256, 256, 0, stream>>>(x, xh, total4);
    }
    {   // decode bias
        decode_bias_kernel<<<(N + 255) / 256, 256, 0, stream>>>(
            bbias, bsign, biasscale, bias, N);
    }
    {   // WMMA GEMM: block tile 128(M) x 256(N), 8 waves of 64x64
        dim3 grid(N / 256, M / 128);
        bitlinear_wmma_kernel<<<grid, 256, 0, stream>>>(xh, Wh, bias, out, M, N, K);
    }
}
